// vGINMolEncoder_15556371546341
// MI455X (gfx1250) — matrix operand order repaired and checked by reading, not compiled, not run on hardware
//
#include <hip/hip_runtime.h>
#include <hip/hip_bf16.h>

typedef __attribute__((ext_vector_type(16))) __bf16 v16bf;
typedef __attribute__((ext_vector_type(8)))  __bf16 v8bf;
typedef __attribute__((ext_vector_type(8)))  float  v8f;

__device__ __forceinline__ unsigned short f2bf(float f) {
  unsigned u = __float_as_uint(f);
  unsigned r = u + 0x7FFFu + ((u >> 16) & 1u);   // round-to-nearest-even
  return (unsigned short)(r >> 16);
}

// ---------------- elementwise / gather-scatter kernels ----------------

__global__ void atom_enc_kernel(const int* __restrict__ x, const float* __restrict__ aemb,
                                float* __restrict__ act, int Nn, int d) {
  int n = blockIdx.x;
  if (n >= Nn) return;
  int xi[9];
#pragma unroll
  for (int f = 0; f < 9; ++f) xi[f] = x[n * 9 + f];
  float* o = act + (size_t)n * d;
  for (int c = threadIdx.x; c < d; c += blockDim.x) {
    float s = 0.f;
#pragma unroll
    for (int f = 0; f < 9; ++f) s += aemb[((size_t)(f * 64 + xi[f])) * d + c];
    o[c] = s;
  }
}

__global__ void vfeat_init_kernel(const float* __restrict__ vn, float* __restrict__ vfeat,
                                  int total, int d) {
  int i = blockIdx.x * blockDim.x + threadIdx.x;
  if (i < total) vfeat[i] = vn[i % d];
}

// act[n] += (usev ? vfeat[batch[n]] : 0); z[n] = act[n]
__global__ void addv_copy_kernel(float* __restrict__ act, float* __restrict__ z,
                                 const float* __restrict__ vfeat, const int* __restrict__ batch,
                                 int Nn, int d, int usev) {
  int n = blockIdx.x;
  if (n >= Nn) return;
  const float* vf = usev ? (vfeat + (size_t)batch[n] * d) : nullptr;
  float* a = act + (size_t)n * d;
  float* zz = z + (size_t)n * d;
  for (int c = threadIdx.x; c < d; c += blockDim.x) {
    float t = a[c] + (usev ? vf[c] : 0.f);
    a[c] = t;
    zz[c] = t;
  }
}

// per-edge: z[dst] += relu(act[src] + bond_emb(edge_attr))
__global__ void edge_msg_kernel(const float* __restrict__ act, float* __restrict__ z,
                                const int* __restrict__ ei, const int* __restrict__ ea,
                                const float* __restrict__ bemb, int E, int d) {
  int e = blockIdx.x;
  if (e >= E) return;
  int s = ei[e], t = ei[E + e];
  int a0 = ea[e * 3 + 0], a1 = ea[e * 3 + 1], a2 = ea[e * 3 + 2];
  const float* b0 = bemb + (size_t)(0 * 16 + a0) * d;
  const float* b1 = bemb + (size_t)(1 * 16 + a1) * d;
  const float* b2 = bemb + (size_t)(2 * 16 + a2) * d;
  const float* hs = act + (size_t)s * d;
  float* zd = z + (size_t)t * d;
  for (int c = threadIdx.x; c < d; c += blockDim.x) {
    float m = hs[c] + b0[c] + b1[c] + b2[c];
    m = m > 0.f ? m : 0.f;
    __hip_atomic_fetch_add(&zd[c], m, __ATOMIC_RELAXED, __HIP_MEMORY_SCOPE_AGENT);
  }
}

// fp32 [rows x cin] -> bf16 [rows x cpad], zero K-padding
__global__ void cvt_pad_kernel(const float* __restrict__ in, unsigned short* __restrict__ out,
                               int rows, int cin, int cpad) {
  int i = blockIdx.x * blockDim.x + threadIdx.x;
  if (i >= rows * cpad) return;
  int r = i / cpad, c = i - r * cpad;
  out[i] = (c < cin) ? f2bf(in[(size_t)r * cin + c]) : (unsigned short)0;
}

// bf16(pooled + vfeat) with K-padding
__global__ void vin_kernel(const float* __restrict__ pooled, const float* __restrict__ vfeat,
                           unsigned short* __restrict__ out, int rows, int cin, int cpad) {
  int i = blockIdx.x * blockDim.x + threadIdx.x;
  if (i >= rows * cpad) return;
  int r = i / cpad, c = i - r * cpad;
  out[i] = (c < cin) ? f2bf(pooled[(size_t)r * cin + c] + vfeat[(size_t)r * cin + c])
                     : (unsigned short)0;
}

__global__ void pool_kernel(const float* __restrict__ act, float* __restrict__ pooled,
                            const int* __restrict__ batch, int Nn, int d) {
  int n = blockIdx.x;
  if (n >= Nn) return;
  int g = batch[n];
  const float* a = act + (size_t)n * d;
  float* p = pooled + (size_t)g * d;
  for (int c = threadIdx.x; c < d; c += blockDim.x)
    __hip_atomic_fetch_add(&p[c], a[c], __ATOMIC_RELAXED, __HIP_MEMORY_SCOPE_AGENT);
}

__global__ void readout_acc_kernel(const float* __restrict__ act, float* __restrict__ out,
                                   int* __restrict__ counts, const int* __restrict__ batch,
                                   int Nn, int d) {
  int n = blockIdx.x;
  if (n >= Nn) return;
  int g = batch[n];
  if (threadIdx.x == 0)
    __hip_atomic_fetch_add(&counts[g], 1, __ATOMIC_RELAXED, __HIP_MEMORY_SCOPE_AGENT);
  const float* a = act + (size_t)n * d;
  float* o = out + (size_t)g * d;
  for (int c = threadIdx.x; c < d; c += blockDim.x)
    __hip_atomic_fetch_add(&o[c], a[c], __ATOMIC_RELAXED, __HIP_MEMORY_SCOPE_AGENT);
}

__global__ void readout_div_kernel(float* __restrict__ out, const int* __restrict__ counts,
                                   int total, int d) {
  int i = blockIdx.x * blockDim.x + threadIdx.x;
  if (i >= total) return;
  float cnt = (float)counts[i / d];
  out[i] = out[i] / fmaxf(cnt, 1.0f);
}

// ---------------- weight / BN preparation ----------------

// W [K x Nc] fp32 row-major -> Wt bf16 [Npad x Kpad] (Wt[n,k] = W[k,n]), zero padded
__global__ void wtrans_kernel(const float* __restrict__ W, unsigned short* __restrict__ Wt,
                              int K, int Nc, int Kpad, int Npad) {
  int i = blockIdx.x * blockDim.x + threadIdx.x;
  if (i >= Npad * Kpad) return;
  int n = i / Kpad, k = i - n * Kpad;
  Wt[i] = (n < Nc && k < K) ? f2bf(W[(size_t)k * Nc + n]) : (unsigned short)0;
}

// fold bias + BN (eval) into per-column scale/shift: y = acc*cs + ch
__global__ void bnprep_kernel(const float* __restrict__ bn, const float* __restrict__ bias,
                              float* __restrict__ cs, float* __restrict__ ch, int n, int npad) {
  int i = blockIdx.x * blockDim.x + threadIdx.x;
  if (i >= npad) return;
  if (i < n) {
    float g = bn[0 * n + i], b = bn[1 * n + i], m = bn[2 * n + i], v = bn[3 * n + i];
    float sc = g * rsqrtf(v + 1e-5f);
    cs[i] = sc;
    ch[i] = b - m * sc + bias[i] * sc;
  } else {
    cs[i] = 0.f;
    ch[i] = 0.f;
  }
}

// ---------------- WMMA bf16 GEMM: async global->LDS B staging ----------------
// C[M x N] = A[M x Kpad](bf16, row-major) * Bt[Npad x Kpad](bf16, B-column-major)
// Block: 8 waves, 128x64 tile. B tile (64 cols x 64 K) staged in LDS via
// global_load_async_to_lds_b128 (ASYNCcnt), double-buffered; all 8 waves share it.
// LDS column stride = 72 shorts (144B) => conflict-free ds_load_b128 fragments.
// Per K-step: all 4 B fragments loaded first (one s_wait_dscnt), then 4
// back-to-back WMMAs on independent accumulators.

#define BCOL_STRIDE 72   // shorts per column in LDS tile (64 data + 8 pad)

__device__ __forceinline__ void async_copy_b_tile(const unsigned short* __restrict__ Bt,
                                                  int nt0, int Kpad, int ks,
                                                  unsigned short* bufptr, int tid) {
#pragma unroll
  for (int i = 0; i < 2; ++i) {
    int idx = tid + i * 256;          // 512 x 16B segments = 64 cols x 128B
    int col = idx >> 3;
    int seg = idx & 7;
    const unsigned short* gp = Bt + (size_t)(nt0 + col) * Kpad + ks + seg * 8;
    unsigned ldsoff = (unsigned)(uintptr_t)(bufptr + col * BCOL_STRIDE + seg * 8);
    asm volatile("global_load_async_to_lds_b128 %0, %1, off"
                 :: "v"(ldsoff), "v"(gp)
                 : "memory");
  }
}

// B fragment: 16 contiguous bf16 (two ds_load_b128 from LDS)
__device__ __forceinline__ v16bf frag_b(const unsigned short* p) {
  v8bf lo = *(const v8bf*)p;
  v8bf hi = *(const v8bf*)(p + 8);
  v16bf r;
#pragma unroll
  for (int i = 0; i < 8; ++i) { r[i] = lo[i]; r[8 + i] = hi[i]; }
  return r;
}

// A fragment: 16 contiguous bf16 (elems 0..7 at +0, 8..15 at +8)
__device__ __forceinline__ v16bf frag_a(const unsigned short* p) {
  v8bf lo = *(const v8bf*)p;
  v8bf hi = *(const v8bf*)(p + 8);
  v16bf r;
#pragma unroll
  for (int i = 0; i < 8; ++i) { r[i] = lo[i]; r[8 + i] = hi[i]; }
  return r;
}

__global__ __launch_bounds__(256)
void gemm_bf16_kernel(const unsigned short* __restrict__ A,
                      const unsigned short* __restrict__ Bt,
                      const float* __restrict__ cs, const float* __restrict__ ch,
                      int M, int Ncols, int Kpad,
                      unsigned short* __restrict__ outb, float* __restrict__ outf,
                      int ldo, int relu) {
  __shared__ __align__(16) unsigned short Bs[2][64 * BCOL_STRIDE];
  const int tid = threadIdx.x;
  const int lane = tid & 31;
  const int wave = tid >> 5;
  const int mt = blockIdx.x * 128 + wave * 16;
  const int nt0 = blockIdx.y * 64;

  // A fragment: lane holds row M = lane&15; 16 consecutive K from (lane&16)?16:0
  int mrow = mt + (lane & 15);
  int rowc = mrow < M ? mrow : (M - 1);            // clamp; clamped rows never stored
  const unsigned short* arow = A + (size_t)rowc * Kpad + ((lane & 16) ? 16 : 0);

  // B fragment: lane holds col N = lane&15; 16 consecutive K from (lane&16)?16:0
  const int coll = lane & 15;
  const int bko = (lane & 16) ? 16 : 0;
  const int nk = Kpad >> 6;                        // K chunks of 64 (Kpad % 64 == 0)

  async_copy_b_tile(Bt, nt0, Kpad, 0, &Bs[0][0], tid);

  v8f acc[4] = {};
  for (int kc = 0; kc < nk; ++kc) {
    const int ks = kc << 6;
    // A fragments for both K-steps of this chunk (global; latency hides under wait+barrier)
    v16bf a0 = frag_a(arow + ks);
    v16bf a1 = frag_a(arow + ks + 32);
    __builtin_prefetch(arow + ks + 64, 0, 0);      // global_prefetch_b8 next A slab

    if (kc + 1 < nk) {
      async_copy_b_tile(Bt, nt0, Kpad, ks + 64, &Bs[(kc + 1) & 1][0], tid);
      asm volatile("s_wait_asynccnt 0x2" ::: "memory");  // own chunk-kc DMAs done (in-order)
    } else {
      asm volatile("s_wait_asynccnt 0x0" ::: "memory");
    }
    __syncthreads();                               // all waves' DMA writes visible

    const unsigned short* bb = &Bs[kc & 1][0] + coll * BCOL_STRIDE + bko;
    v16bf bf[4];
    // K-step ks+0..31: load all 4 B fragments, then 4 independent WMMAs
#pragma unroll
    for (int j = 0; j < 4; ++j) bf[j] = frag_b(bb + j * (16 * BCOL_STRIDE));
#pragma unroll
    for (int j = 0; j < 4; ++j)
      acc[j] = __builtin_amdgcn_wmma_f32_16x16x32_bf16(
          false, a0, false, bf[j], (short)0, acc[j], false, false);
    // K-step ks+32..63
#pragma unroll
    for (int j = 0; j < 4; ++j) bf[j] = frag_b(bb + j * (16 * BCOL_STRIDE) + 32);
#pragma unroll
    for (int j = 0; j < 4; ++j)
      acc[j] = __builtin_amdgcn_wmma_f32_16x16x32_bf16(
          false, a1, false, bf[j], (short)0, acc[j], false, false);
    __syncthreads();                               // protect buffer before next overwrite
  }

  // C/D layout: VGPR r -> M = r (+8 for lanes 16-31), N = lane&15
  const int rbase = mt + ((lane & 16) ? 8 : 0);
#pragma unroll
  for (int j = 0; j < 4; ++j) {
#pragma unroll
    for (int r = 0; r < 8; ++r) {
      int orow = rbase + r;
      if (orow >= M) continue;
      int col = nt0 + j * 16 + coll;
      float y = acc[j][r] * cs[col] + ch[col];     // bias+BN folded
      if (relu) y = fmaxf(y, 0.f);
      if (outb) {
        outb[(size_t)orow * ldo + col] = f2bf(y);  // padded cols give y==0
      } else if (col < Ncols) {
        outf[(size_t)orow * ldo + col] = y;
      }
    }
  }
}

// ---------------- host orchestration ----------------

extern "C" void kernel_launch(void* const* d_in, const int* in_sizes, int n_in,
                              void* d_out, int out_size, void* d_ws, size_t ws_size,
                              hipStream_t stream) {
  const int d   = in_sizes[12];          // 300  (vn_emb = [1,d])
  const int d2  = 2 * d;                 // 600
  const int Nn  = in_sizes[3];           // 200000
  const int E   = in_sizes[1] / 2;       // 800000
  const int G_  = out_size / d;          // 8000
  const int Lh  = in_sizes[7] / d2;      // 5
  const int Kp1 = 64 * ((d  + 63) / 64); // 320 : padded K for d-wide inputs
  const int Kp2 = 64 * ((d2 + 63) / 64); // 640 : padded K for 2d-wide inputs
  const int Np1 = Kp2;                   // padded N for 600-col GEMMs (64-tiles)
  const int Np2 = Kp1;                   // padded N for 300-col GEMMs

  const int*   x     = (const int*)d_in[0];
  const int*   ei    = (const int*)d_in[1];
  const int*   ea    = (const int*)d_in[2];
  const int*   batch = (const int*)d_in[3];
  const float* aemb  = (const float*)d_in[4];
  const float* bemb  = (const float*)d_in[5];
  const float* W1    = (const float*)d_in[6];
  const float* b1    = (const float*)d_in[7];
  const float* bnin  = (const float*)d_in[8];
  const float* W2    = (const float*)d_in[9];
  const float* b2    = (const float*)d_in[10];
  const float* bnout = (const float*)d_in[11];
  const float* vn    = (const float*)d_in[12];
  const float* vW1   = (const float*)d_in[13];
  const float* vb1   = (const float*)d_in[14];
  const float* vbn1  = (const float*)d_in[15];
  const float* vW2   = (const float*)d_in[16];
  const float* vb2   = (const float*)d_in[17];
  const float* vbn2  = (const float*)d_in[18];

  char* p = (char*)d_ws;
  auto alloc = [&](size_t bytes) -> char* {
    char* r = p;
    p += (bytes + 255) & ~(size_t)255;
    return r;
  };
  float*          act    = (float*)alloc((size_t)Nn * d * 4);
  float*          zf     = (float*)alloc((size_t)Nn * d * 4);
  unsigned short* zb     = (unsigned short*)alloc((size_t)Nn * Kp1 * 2);
  unsigned short* tb     = (unsigned short*)alloc((size_t)Nn * Kp2 * 2);
  float*          pooled = (float*)alloc((size_t)G_ * d * 4);
  float*          vfeat  = (float*)alloc((size_t)G_ * d * 4);
  unsigned short* vinb   = (unsigned short*)alloc((size_t)G_ * Kp1 * 2);
  unsigned short* vtb    = (unsigned short*)alloc((size_t)G_ * Kp2 * 2);
  int*            counts = (int*)alloc((size_t)G_ * 4);
  unsigned short* W1t    = (unsigned short*)alloc((size_t)Lh * Np1 * Kp1 * 2);
  unsigned short* W2t    = (unsigned short*)alloc((size_t)Lh * Np2 * Kp2 * 2);
  unsigned short* vW1t   = (unsigned short*)alloc((size_t)Np1 * Kp1 * 2);
  unsigned short* vW2t   = (unsigned short*)alloc((size_t)Np2 * Kp2 * 2);
  float* cs1  = (float*)alloc((size_t)Lh * Np1 * 4);
  float* ch1  = (float*)alloc((size_t)Lh * Np1 * 4);
  float* cs2  = (float*)alloc((size_t)Lh * Np2 * 4);
  float* ch2  = (float*)alloc((size_t)Lh * Np2 * 4);
  float* vcs1 = (float*)alloc((size_t)Np1 * 4);
  float* vch1 = (float*)alloc((size_t)Np1 * 4);
  float* vcs2 = (float*)alloc((size_t)Np2 * 4);
  float* vch2 = (float*)alloc((size_t)Np2 * 4);

  // ---- weight transpose + bf16 convert, BN folding ----
  int wtg1 = (Np1 * Kp1 + 255) / 256, wtg2 = (Np2 * Kp2 + 255) / 256;
  for (int l = 0; l < Lh; ++l) {
    wtrans_kernel<<<wtg1, 256, 0, stream>>>(W1 + (size_t)l * d * d2,
                                            W1t + (size_t)l * Np1 * Kp1, d, d2, Kp1, Np1);
    wtrans_kernel<<<wtg2, 256, 0, stream>>>(W2 + (size_t)l * d2 * d,
                                            W2t + (size_t)l * Np2 * Kp2, d2, d, Kp2, Np2);
    bnprep_kernel<<<(Np1 + 63) / 64, 64, 0, stream>>>(bnin + (size_t)l * 4 * d2, b1 + (size_t)l * d2,
                                                      cs1 + l * Np1, ch1 + l * Np1, d2, Np1);
    bnprep_kernel<<<(Np2 + 63) / 64, 64, 0, stream>>>(bnout + (size_t)l * 4 * d, b2 + (size_t)l * d,
                                                      cs2 + l * Np2, ch2 + l * Np2, d, Np2);
  }
  wtrans_kernel<<<wtg1, 256, 0, stream>>>(vW1, vW1t, d, d2, Kp1, Np1);
  wtrans_kernel<<<wtg2, 256, 0, stream>>>(vW2, vW2t, d2, d, Kp2, Np2);
  bnprep_kernel<<<(Np1 + 63) / 64, 64, 0, stream>>>(vbn1, vb1, vcs1, vch1, d2, Np1);
  bnprep_kernel<<<(Np2 + 63) / 64, 64, 0, stream>>>(vbn2, vb2, vcs2, vch2, d, Np2);

  // ---- encoders ----
  atom_enc_kernel<<<Nn, 128, 0, stream>>>(x, aemb, act, Nn, d);
  vfeat_init_kernel<<<(G_ * d + 255) / 256, 256, 0, stream>>>(vn, vfeat, G_ * d, d);

  dim3 gGemm1((Nn + 127) / 128, Np1 / 64);
  dim3 gGemm2((Nn + 127) / 128, Np2 / 64);
  dim3 gVG1((G_ + 127) / 128, Np1 / 64);
  dim3 gVG2((G_ + 127) / 128, Np2 / 64);
  int cvtg = (Nn * Kp1 + 255) / 256;
  int vcvtg = (G_ * Kp1 + 255) / 256;

  // ---- layers ----
  for (int l = 0; l < Lh; ++l) {
    addv_copy_kernel<<<Nn, 128, 0, stream>>>(act, zf, vfeat, batch, Nn, d, l > 0 ? 1 : 0);
    edge_msg_kernel<<<E, 128, 0, stream>>>(act, zf, ei, ea, bemb + (size_t)l * 3 * 16 * d, E, d);
    cvt_pad_kernel<<<cvtg, 256, 0, stream>>>(zf, zb, Nn, d, Kp1);
    // inner MLP GEMM1: z @ W1 -> relu(bn_in(.)) -> bf16
    gemm_bf16_kernel<<<gGemm1, 256, 0, stream>>>(zb, W1t + (size_t)l * Np1 * Kp1,
                                                 cs1 + l * Np1, ch1 + l * Np1,
                                                 Nn, d2, Kp1, tb, nullptr, Kp2, 1);
    // inner MLP GEMM2: t @ W2 -> bn_out(.) (+relu except last layer) -> fp32 act
    gemm_bf16_kernel<<<gGemm2, 256, 0, stream>>>(tb, W2t + (size_t)l * Np2 * Kp2,
                                                 cs2 + l * Np2, ch2 + l * Np2,
                                                 Nn, d, Kp2, nullptr, act, d,
                                                 (l != Lh - 1) ? 1 : 0);
    if (l >= 1 && l <= Lh - 2) {  // virtual-node update after layers 1..3
      hipMemsetAsync(pooled, 0, (size_t)G_ * d * 4, stream);
      pool_kernel<<<Nn, 128, 0, stream>>>(act, pooled, batch, Nn, d);
      vin_kernel<<<vcvtg, 256, 0, stream>>>(pooled, vfeat, vinb, G_, d, Kp1);
      gemm_bf16_kernel<<<gVG1, 256, 0, stream>>>(vinb, vW1t, vcs1, vch1,
                                                 G_, d2, Kp1, vtb, nullptr, Kp2, 1);
      gemm_bf16_kernel<<<gVG2, 256, 0, stream>>>(vtb, vW2t, vcs2, vch2,
                                                 G_, d, Kp2, nullptr, vfeat, d, 1);
    }
  }

  // ---- mean-pool readout ----
  hipMemsetAsync(d_out, 0, (size_t)G_ * d * 4, stream);
  hipMemsetAsync(counts, 0, (size_t)G_ * 4, stream);
  readout_acc_kernel<<<Nn, 128, 0, stream>>>(act, (float*)d_out, counts, batch, Nn, d);
  readout_div_kernel<<<(G_ * d + 255) / 256, 256, 0, stream>>>((float*)d_out, counts, G_ * d, d);
}
